// GNNDecoder_62569083568895
// MI455X (gfx1250) — matrix operand order repaired and checked by reading, not compiled, not run on hardware
//
#include <hip/hip_runtime.h>
#include <cstdint>

// ---------------------------------------------------------------------------
// GNN decoder for MI455X (gfx1250): BF16 WMMA (v_wmma_f32_16x16x32_bf16) path.
// Edge MLP = 48 GFLOP dominates -> compute-bound on the matrix cores.
// Messenger kernel: waves split over OUTPUT COLUMNS, each wave covers all 32
// edge rows -> every B fragment feeds 2 WMMAs (2x arithmetic intensity per B
// byte vs row-split), halving L2->VGPR weight traffic so the WMMA pipe, not
// the B stream, is the limiter. A frags: two ds_load_b128 from bf16 LDS tiles;
// B frags: two 16B global loads from N-major (pre-transposed) bf16 weights
// resident in the 192MB L2. K-reduction via wave32 shfl_xor (no LDS atomics).
// ---------------------------------------------------------------------------

#define ZB   4
#define NN   2048
#define KNB  30
#define DE   128
#define DV   128
#define DIN  512   // 2*De + 2*Dv

typedef __attribute__((ext_vector_type(16))) __bf16 v16bf;
typedef __attribute__((ext_vector_type(8)))  float  v8f;

union Frag16 { uint32_t u[8]; v16bf v; };
union AccF   { float f[8];    v8f  v; };

__device__ __forceinline__ uint16_t f2bf(float x) {
    uint32_t u = __float_as_uint(x);
    uint32_t r = (u + 0x7fffu + ((u >> 16) & 1u)) >> 16;   // round-to-nearest-even
    return (uint16_t)r;
}

__device__ __forceinline__ float gelu_exact(float x) {
    return 0.5f * x * (1.0f + erff(x * 0.70710678118654752f));
}

__device__ __forceinline__ v8f vzero8() {
    v8f v;
    #pragma unroll
    for (int i = 0; i < 8; ++i) v[i] = 0.0f;
    return v;
}

// A-matrix fragment (16x32 bf16), row-major source, row = lane&15.
// ISA striping: VGPR v in 0..3 -> K = g*8 + 2v + h ; v in 4..7 -> K = 16 + g*8 + 2(v-4) + h
// => two contiguous 16-byte loads at +g*8 and +16+g*8 halfwords.
__device__ __forceinline__ v16bf load_a_frag(const uint16_t* rowbase, int g) {
    Frag16 f;
    *(uint4*)&f.u[0] = *(const uint4*)(rowbase + g * 8);
    *(uint4*)&f.u[4] = *(const uint4*)(rowbase + 16 + g * 8);
    return f.v;
}

// B-matrix fragment (32x16 bf16), source is N-major transposed weights (BT[n][K]).
// ISA striping (same as A): VGPR v in 0..3 -> K = g*8 + 2v + h ; v in 4..7 -> K = 16 + g*8 + 2(v-4) + h
// => two contiguous 16-byte loads at +g*8 and +16+g*8 halfwords of column n.
__device__ __forceinline__ v16bf load_b_frag(const uint16_t* colbase, int g) {
    Frag16 f;
    *(uint4*)&f.u[0] = *(const uint4*)(colbase + g * 8);
    *(uint4*)&f.u[4] = *(const uint4*)(colbase + 16 + g * 8);
    return f.v;
}

#define WMMA_BF16(a, b, c) \
    __builtin_amdgcn_wmma_f32_16x16x32_bf16(false, (a), false, (b), (short)0, (c), false, false)

// ---------------------------------------------------------------------------
// Kernel 0: fp32 [K][N] weights -> bf16 N-major [N][K] in workspace.
// ---------------------------------------------------------------------------
__global__ void wconv_kernel(const float* __restrict__ src, uint16_t* __restrict__ dst,
                             int K, int Ncols) {
    int idx = blockIdx.x * blockDim.x + threadIdx.x;
    if (idx >= K * Ncols) return;
    int k = idx / Ncols, c = idx - k * Ncols;
    dst[c * K + k] = f2bf(src[idx]);
}

// ---------------------------------------------------------------------------
// Kernel 1: per-node edge messenger. 1 block (64 thr = 2 waves) per node.
// Wave w owns output columns [w*64, w*64+64) across ALL 32 edge rows
// (30 real + 2 zero-pad), so each B fragment is reused for both row tiles.
// ---------------------------------------------------------------------------
__global__ __launch_bounds__(64) void messenger_kernel(
    const float* __restrict__ V_new, const float* __restrict__ V_old,
    const float* __restrict__ Eft,   const float* __restrict__ S,
    const int*   __restrict__ K_idx,
    const unsigned char* __restrict__ edge_mask,
    const unsigned char* __restrict__ ar_mask,
    const uint16_t* __restrict__ W0T, const uint16_t* __restrict__ W1T,
    const uint16_t* __restrict__ W2T,
    const float* __restrict__ b_m0, const float* __restrict__ b_m1,
    const float* __restrict__ b_m2,
    const float* __restrict__ g_mn, const float* __restrict__ beta_mn,
    float* __restrict__ V_mid)
{
    __shared__ __align__(16) uint16_t As[32 * DIN];   // 32 KB gathered inputs (bf16)
    __shared__ __align__(16) uint16_t H1[32 * DV];    // 8 KB layer0 activations
    __shared__ __align__(16) uint16_t H2[32 * DV];    // 8 KB layer1 activations
    __shared__ float Vacc[DV];
    __shared__ float emask[32];
    __shared__ float red[64], red2[64];

    const int node = blockIdx.x;
    const int z = node / NN, n = node - z * NN;
    const int tid  = threadIdx.x;
    const int lane = tid & 31;
    const int wv   = tid >> 5;          // wave id: column half  [wv*64, wv*64+64)
    const int g    = lane >> 4;         // half-wave
    const int mm   = lane & 15;         // A row within row tile
    const int nn   = lane & 15;         // B/C column within col tile

    // ---- stage A: gather+concat [Vi | Vj | E | Es] -> 32x512 bf16 ----
    {
        const int r  = tid >> 1;              // row 0..31
        const int c0 = (tid & 1) * 256;       // half-row
        uint16_t* dst = &As[r * DIN];
        if (r < KNB) {
            const int eoff = (z * NN + n) * KNB + r;
            const int j    = K_idx[eoff];
            const bool am  = ar_mask[eoff] != 0;
            const float* vi = V_new + (size_t)(z * NN + n) * DV;
            const float* vj = (am ? V_new : V_old) + (size_t)(z * NN + j) * DV;
            const float* er = Eft + (size_t)eoff * DE;
            const float* sr = S + (size_t)(z * NN + j) * DE;
            #pragma unroll 4
            for (int c = c0; c < c0 + 256; c += 4) {
                float4 x;
                if (c < 128)      x = *(const float4*)(vi + c);
                else if (c < 256) x = *(const float4*)(vj + (c - 128));
                else if (c < 384) x = *(const float4*)(er + (c - 256));
                else              x = am ? *(const float4*)(sr + (c - 384))
                                         : make_float4(0.f, 0.f, 0.f, 0.f);
                uint32_t p0 = (uint32_t)f2bf(x.x) | ((uint32_t)f2bf(x.y) << 16);
                uint32_t p1 = (uint32_t)f2bf(x.z) | ((uint32_t)f2bf(x.w) << 16);
                *(uint2*)&dst[c] = make_uint2(p0, p1);
            }
        } else {
            for (int c = c0; c < c0 + 256; c += 4)
                *(uint2*)&dst[c] = make_uint2(0u, 0u);
        }
    }
    if (tid < 32)
        emask[tid] = (tid < KNB && edge_mask[(z * NN + n) * KNB + tid]) ? 1.0f : 0.0f;
    // warm WGP$ with this wave's weight columns while staging drains
    #pragma unroll
    for (int j = 0; j < 4; ++j) {
        const int col = (wv * 4 + j) * 16 + nn;
        __builtin_prefetch(W0T + col * DIN + g * 256, 0, 3);   // global_prefetch_b8
        __builtin_prefetch(W1T + col * DV, 0, 3);
        __builtin_prefetch(W2T + col * DV, 0, 3);
    }
    __syncthreads();

    v8f acc[2][4];   // [row tile][col tile within wave]

    // ---- layer 0: 32x512 @ 512x128 (16 k-steps, 4 col-tiles x 2 row-tiles) ----
    #pragma unroll
    for (int rt = 0; rt < 2; ++rt)
        #pragma unroll
        for (int j = 0; j < 4; ++j) acc[rt][j] = vzero8();
    {
        for (int ks = 0; ks < 16; ++ks) {
            v16bf a0 = load_a_frag(&As[mm * DIN + ks * 32], g);
            v16bf a1 = load_a_frag(&As[(16 + mm) * DIN + ks * 32], g);
            #pragma unroll
            for (int j = 0; j < 4; ++j) {
                v16bf b = load_b_frag(W0T + ((wv * 4 + j) * 16 + nn) * DIN + ks * 32, g);
                acc[0][j] = WMMA_BF16(a0, b, acc[0][j]);
                acc[1][j] = WMMA_BF16(a1, b, acc[1][j]);
            }
        }
        #pragma unroll
        for (int j = 0; j < 4; ++j) {
            const int col = (wv * 4 + j) * 16 + nn;
            const float bias = b_m0[col];
            #pragma unroll
            for (int rt = 0; rt < 2; ++rt) {
                AccF t; t.v = acc[rt][j];
                #pragma unroll
                for (int r = 0; r < 8; ++r)
                    H1[(rt * 16 + g * 8 + r) * DV + col] = f2bf(gelu_exact(t.f[r] + bias));
            }
        }
    }
    __syncthreads();

    // ---- layer 1: 32x128 @ 128x128 ----
    #pragma unroll
    for (int rt = 0; rt < 2; ++rt)
        #pragma unroll
        for (int j = 0; j < 4; ++j) acc[rt][j] = vzero8();
    {
        for (int ks = 0; ks < 4; ++ks) {
            v16bf a0 = load_a_frag(&H1[mm * DV + ks * 32], g);
            v16bf a1 = load_a_frag(&H1[(16 + mm) * DV + ks * 32], g);
            #pragma unroll
            for (int j = 0; j < 4; ++j) {
                v16bf b = load_b_frag(W1T + ((wv * 4 + j) * 16 + nn) * DV + ks * 32, g);
                acc[0][j] = WMMA_BF16(a0, b, acc[0][j]);
                acc[1][j] = WMMA_BF16(a1, b, acc[1][j]);
            }
        }
        #pragma unroll
        for (int j = 0; j < 4; ++j) {
            const int col = (wv * 4 + j) * 16 + nn;
            const float bias = b_m1[col];
            #pragma unroll
            for (int rt = 0; rt < 2; ++rt) {
                AccF t; t.v = acc[rt][j];
                #pragma unroll
                for (int r = 0; r < 8; ++r)
                    H2[(rt * 16 + g * 8 + r) * DV + col] = f2bf(gelu_exact(t.f[r] + bias));
            }
        }
    }
    __syncthreads();

    // ---- layer 2: 32x128 @ 128x128, edge-mask + reduce over K ----
    #pragma unroll
    for (int rt = 0; rt < 2; ++rt)
        #pragma unroll
        for (int j = 0; j < 4; ++j) acc[rt][j] = vzero8();
    {
        for (int ks = 0; ks < 4; ++ks) {
            v16bf a0 = load_a_frag(&H2[mm * DV + ks * 32], g);
            v16bf a1 = load_a_frag(&H2[(16 + mm) * DV + ks * 32], g);
            #pragma unroll
            for (int j = 0; j < 4; ++j) {
                v16bf b = load_b_frag(W2T + ((wv * 4 + j) * 16 + nn) * DV + ks * 32, g);
                acc[0][j] = WMMA_BF16(a0, b, acc[0][j]);
                acc[1][j] = WMMA_BF16(a1, b, acc[1][j]);
            }
        }
        #pragma unroll
        for (int j = 0; j < 4; ++j) {
            const int col = (wv * 4 + j) * 16 + nn;
            const float bias = b_m2[col];
            float s = 0.0f;
            #pragma unroll
            for (int rt = 0; rt < 2; ++rt) {
                AccF t; t.v = acc[rt][j];
                #pragma unroll
                for (int r = 0; r < 8; ++r)
                    s += (t.f[r] + bias) * emask[rt * 16 + g * 8 + r];
            }
            // combine the two half-wave row groups; each column owned by one lane
            s += __shfl_xor(s, 16, 32);
            if (g == 0) Vacc[col] = s;
        }
    }
    __syncthreads();

    // ---- residual + LayerNorm -> V_mid (64 threads x 2 cols) ----
    const float* vrow = V_new + (size_t)(z * NN + n) * DV;
    float y0 = vrow[tid]      + Vacc[tid];
    float y1 = vrow[tid + 64] + Vacc[tid + 64];
    red[tid]  = y0 + y1;
    red2[tid] = y0 * y0 + y1 * y1;
    __syncthreads();
    for (int off = 32; off > 0; off >>= 1) {
        if (tid < off) { red[tid] += red[tid + off]; red2[tid] += red2[tid + off]; }
        __syncthreads();
    }
    const float mean = red[0] * (1.0f / DV);
    const float var  = red2[0] * (1.0f / DV) - mean * mean;
    const float rstd = rsqrtf(var + 1e-5f);
    float* out = V_mid + (size_t)(z * NN + n) * DV;
    out[tid]      = (y0 - mean) * rstd * g_mn[tid]      + beta_mn[tid];
    out[tid + 64] = (y1 - mean) * rstd * g_mn[tid + 64] + beta_mn[tid + 64];
}

// ---------------------------------------------------------------------------
// Kernel 2: FFN over nodes. 1 block (128 thr = 4 waves) per 16-node row tile.
// Col-tiles are already partitioned across waves (no duplicated B traffic).
// ---------------------------------------------------------------------------
__global__ __launch_bounds__(128) void ffn_kernel(
    const float* __restrict__ V_mid,
    const uint16_t* __restrict__ F0T, const uint16_t* __restrict__ F1T,
    const float* __restrict__ b_f0, const float* __restrict__ b_f1,
    const float* __restrict__ g_fn, const float* __restrict__ beta_fn,
    float* __restrict__ Vout)
{
    __shared__ __align__(16) uint16_t Xs[16 * DV];       // 4 KB inputs (bf16)
    __shared__ __align__(16) uint16_t Hs[16 * 4 * DV];   // 16 KB hidden (bf16)
    __shared__ float Of[16 * DV];                        // 8 KB ffn output (f32)
    __shared__ float pred[128], pred2[128];

    const int row0 = blockIdx.x * 16;
    const int tid  = threadIdx.x;
    const int lane = tid & 31;
    const int wv   = tid >> 5;
    const int g    = lane >> 4;
    const int mm   = lane & 15;
    const int nn   = lane & 15;

    // stage X tile (16 x 128) as bf16
    {
        const int r  = tid >> 3;
        const int cs = (tid & 7) * 16;
        const float* src = V_mid + (size_t)(row0 + r) * DV + cs;
        uint16_t* dst = &Xs[r * DV + cs];
        #pragma unroll
        for (int c = 0; c < 16; c += 4) {
            float4 x = *(const float4*)(src + c);
            uint32_t p0 = (uint32_t)f2bf(x.x) | ((uint32_t)f2bf(x.y) << 16);
            uint32_t p1 = (uint32_t)f2bf(x.z) | ((uint32_t)f2bf(x.w) << 16);
            *(uint2*)&dst[c] = make_uint2(p0, p1);
        }
    }
    __syncthreads();

    // f0: 16x128 @ 128x512 ; wave owns 8 of 32 col-tiles
    {
        v8f acc[8];
        #pragma unroll
        for (int t = 0; t < 8; ++t) acc[t] = vzero8();
        for (int ks = 0; ks < 4; ++ks) {
            v16bf a = load_a_frag(&Xs[mm * DV + ks * 32], g);
            #pragma unroll
            for (int j = 0; j < 8; ++j) {
                const int ct = wv * 8 + j;
                v16bf b = load_b_frag(F0T + (ct * 16 + nn) * DV + ks * 32, g);
                acc[j] = WMMA_BF16(a, b, acc[j]);
            }
        }
        #pragma unroll
        for (int j = 0; j < 8; ++j) {
            const int col = (wv * 8 + j) * 16 + nn;
            const float bias = b_f0[col];
            AccF t; t.v = acc[j];
            #pragma unroll
            for (int r = 0; r < 8; ++r)
                Hs[(g * 8 + r) * (4 * DV) + col] = f2bf(gelu_exact(t.f[r] + bias));
        }
    }
    __syncthreads();

    // f1: 16x512 @ 512x128 ; wave owns 2 of 8 col-tiles
    {
        v8f acc[2];
        acc[0] = vzero8(); acc[1] = vzero8();
        for (int ks = 0; ks < 16; ++ks) {
            v16bf a = load_a_frag(&Hs[mm * (4 * DV) + ks * 32], g);
            #pragma unroll
            for (int j = 0; j < 2; ++j) {
                const int ct = wv * 2 + j;
                v16bf b = load_b_frag(F1T + (ct * 16 + nn) * (4 * DV) + ks * 32, g);
                acc[j] = WMMA_BF16(a, b, acc[j]);
            }
        }
        #pragma unroll
        for (int j = 0; j < 2; ++j) {
            const int col = (wv * 2 + j) * 16 + nn;
            const float bias = b_f1[col];
            AccF t; t.v = acc[j];
            #pragma unroll
            for (int r = 0; r < 8; ++r)
                Of[(g * 8 + r) * DV + col] = t.f[r] + bias;
        }
    }
    __syncthreads();

    // residual + LayerNorm (8 threads per row, 16 cols each)
    const int r  = tid >> 3;
    const int cs = (tid & 7) * 16;
    const float* vm = V_mid + (size_t)(row0 + r) * DV;
    float x[16], s = 0.0f, s2 = 0.0f;
    #pragma unroll
    for (int c = 0; c < 16; ++c) {
        float v = vm[cs + c] + Of[r * DV + cs + c];
        x[c] = v; s += v; s2 += v * v;
    }
    pred[tid] = s; pred2[tid] = s2;
    __syncthreads();
    float ts = 0.0f, ts2 = 0.0f;
    #pragma unroll
    for (int i = 0; i < 8; ++i) { ts += pred[(r << 3) + i]; ts2 += pred2[(r << 3) + i]; }
    const float mean = ts * (1.0f / DV);
    const float var  = ts2 * (1.0f / DV) - mean * mean;
    const float rstd = rsqrtf(var + 1e-5f);
    float* o = Vout + (size_t)(row0 + r) * DV + cs;
    #pragma unroll
    for (int c = 0; c < 16; ++c)
        o[c] = (x[c] - mean) * rstd * g_fn[cs + c] + beta_fn[cs + c];
}

// ---------------------------------------------------------------------------
// Host launcher
// ---------------------------------------------------------------------------
extern "C" void kernel_launch(void* const* d_in, const int* in_sizes, int n_in,
                              void* d_out, int out_size, void* d_ws, size_t ws_size,
                              hipStream_t stream) {
    const float* V_new = (const float*)d_in[0];
    const float* V_old = (const float*)d_in[1];
    const float* Eft   = (const float*)d_in[2];
    const float* S     = (const float*)d_in[3];
    const int*   K_idx = (const int*)d_in[4];
    const unsigned char* edge_mask = (const unsigned char*)d_in[5];
    const unsigned char* ar_mask   = (const unsigned char*)d_in[6];
    const float* W_m0 = (const float*)d_in[7];
    const float* b_m0 = (const float*)d_in[8];
    const float* W_m1 = (const float*)d_in[9];
    const float* b_m1 = (const float*)d_in[10];
    const float* W_m2 = (const float*)d_in[11];
    const float* b_m2 = (const float*)d_in[12];
    const float* g_mn   = (const float*)d_in[13];
    const float* beta_mn= (const float*)d_in[14];
    const float* W_f0 = (const float*)d_in[15];
    const float* b_f0 = (const float*)d_in[16];
    const float* W_f1 = (const float*)d_in[17];
    const float* b_f1 = (const float*)d_in[18];
    const float* g_fn   = (const float*)d_in[19];
    const float* beta_fn= (const float*)d_in[20];

    uint8_t* ws = (uint8_t*)d_ws;
    // workspace layout (256B-aligned)
    uint16_t* W0T = (uint16_t*)(ws + 0);              // 512*128*2  = 131072
    uint16_t* W1T = (uint16_t*)(ws + 131072);         // 128*128*2  =  32768
    uint16_t* W2T = (uint16_t*)(ws + 163840);         // 128*128*2  =  32768
    uint16_t* F0T = (uint16_t*)(ws + 196608);         // 128*512*2  = 131072
    uint16_t* F1T = (uint16_t*)(ws + 327680);         // 512*128*2  = 131072
    float*  V_mid = (float*)   (ws + 458752);         // Z*N*128*4  = 4 MB

    // weight convert + transpose (bf16, N-major) -- stays hot in 192MB L2
    wconv_kernel<<<(512 * 128 + 255) / 256, 256, 0, stream>>>(W_m0, W0T, 512, 128);
    wconv_kernel<<<(128 * 128 + 255) / 256, 256, 0, stream>>>(W_m1, W1T, 128, 128);
    wconv_kernel<<<(128 * 128 + 255) / 256, 256, 0, stream>>>(W_m2, W2T, 128, 128);
    wconv_kernel<<<(128 * 512 + 255) / 256, 256, 0, stream>>>(W_f0, F0T, 128, 512);
    wconv_kernel<<<(512 * 128 + 255) / 256, 256, 0, stream>>>(W_f1, F1T, 512, 128);

    messenger_kernel<<<ZB * NN, 64, 0, stream>>>(
        V_new, V_old, Eft, S, K_idx, edge_mask, ar_mask,
        W0T, W1T, W2T, b_m0, b_m1, b_m2, g_mn, beta_mn, V_mid);

    ffn_kernel<<<(ZB * NN) / 16, 128, 0, stream>>>(
        V_mid, F0T, F1T, b_f0, b_f1, g_fn, beta_fn, (float*)d_out);
}
